// MultiHeadAttention_70566312673909
// MI455X (gfx1250) — compile-verified, matrix operand order checked by reading, not run on hardware
//
#include <hip/hip_runtime.h>

#define TT    2048
#define BB    2
#define NEMBD 1024
#define NHEAD 16
#define HDIM  64
#define BT    (BB * TT)   // 4096 rows total

typedef __attribute__((ext_vector_type(16))) __bf16 v16bf;
typedef __attribute__((ext_vector_type(8)))  float  v8f;

union FragU { uint4  u[2]; v16bf v; };
union FragH { __bf16 h[16]; v16bf v; };

// Load a 16x32 bf16 WMMA fragment (A- or B-matrix) from a row-major matrix.
// Per the CDNA5 ISA layout: lane L holds K = k0 + (L/16)*8 + {0..7} in VGPRs
// 0..3 and K = k0 + 16 + (L/16)*8 + {0..7} in VGPRs 4..7 -> two b128 loads.
__device__ __forceinline__ v16bf load_frag_row(const __bf16* __restrict__ row,
                                               int k0, int half) {
  FragU f;
  const __bf16* p = row + k0 + half * 8;
  f.u[0] = *reinterpret_cast<const uint4*>(p);
  f.u[1] = *reinterpret_cast<const uint4*>(p + 16);
  return f.v;
}

__device__ __forceinline__ v8f wmma_bf16(v16bf a, v16bf b, v8f c) {
  return __builtin_amdgcn_wmma_f32_16x16x32_bf16(false, a, false, b,
                                                 (short)0, c, false, false);
}

// One K-chunk stage for the 16x64 GEMM tiles: A fragment + 4 B fragments.
__device__ __forceinline__ void load_group(v16bf& a, v16bf b[4],
                                           const __bf16* __restrict__ arow,
                                           const __bf16* const wr[4],
                                           int k0, int half) {
  a = load_frag_row(arow, k0, half);
#pragma unroll
  for (int j = 0; j < 4; ++j) b[j] = load_frag_row(wr[j], k0, half);
}

__device__ __forceinline__ void mac_group(v8f acc[4], const v16bf& a,
                                          const v16bf b[4]) {
#pragma unroll
  for (int j = 0; j < 4; ++j) acc[j] = wmma_bf16(a, b[j], acc[j]);
}

// ---------------------------------------------------------------- fp32->bf16
__global__ void __launch_bounds__(256) cvt_f32_bf16(const float* __restrict__ in,
                                                    __bf16* __restrict__ out,
                                                    int n4) {
  int i = blockIdx.x * blockDim.x + threadIdx.x;
  if (i >= n4) return;
  float4 f = reinterpret_cast<const float4*>(in)[i];
  union { __bf16 h[4]; uint2 u; } o;
  o.h[0] = (__bf16)f.x; o.h[1] = (__bf16)f.y;
  o.h[2] = (__bf16)f.z; o.h[3] = (__bf16)f.w;
  reinterpret_cast<uint2*>(out)[i] = o.u;
}

// ------------------------------------------------------- fused QKV projection
// grid (16, 64, 3), block 128 (4 waves). Each wave: 16 rows x 64 cols of
// y = x @ W^T + b over K=1024. Ping-pong double buffering: the two fragment
// sets alternate structurally, so no inter-iteration register copies (and no
// WMMA->VALU WAR hazard NOPs) are needed. z selects {Q,K,V}.
__global__ void __launch_bounds__(128) qkv_gemm(
    const __bf16* __restrict__ xb,
    const __bf16* __restrict__ wq, const __bf16* __restrict__ wk,
    const __bf16* __restrict__ wv,
    const float* __restrict__ bq, const float* __restrict__ bk,
    const float* __restrict__ bv,
    __bf16* __restrict__ q, __bf16* __restrict__ kmat,
    __bf16* __restrict__ vt) {
  const int lane = threadIdx.x & 31;
  const int wid  = threadIdx.x >> 5;
  const int half = lane >> 4;
  const int n    = lane & 15;
  const int rowTile = blockIdx.y * 4 + wid;   // 0..255
  const int colGrp  = blockIdx.x;             // 0..15
  const int which   = blockIdx.z;

  const __bf16* W    = (which == 0) ? wq : (which == 1) ? wk : wv;
  const float*  bias = (which == 0) ? bq : (which == 1) ? bk : bv;

  const __bf16* arow = xb + (size_t)(rowTile * 16 + n) * NEMBD;
  const __bf16* wr[4];
#pragma unroll
  for (int j = 0; j < 4; ++j)
    wr[j] = W + (size_t)(colGrp * 64 + j * 16 + n) * NEMBD;

  v8f acc[4] = {};
  v16bf a0, a1, b0[4], b1[4];
  load_group(a0, b0, arow, wr, 0, half);
  for (int k0 = 32; k0 < NEMBD - 32; k0 += 64) {
    load_group(a1, b1, arow, wr, k0, half);
    mac_group(acc, a0, b0);
    load_group(a0, b0, arow, wr, k0 + 32, half);
    mac_group(acc, a1, b1);
  }
  load_group(a1, b1, arow, wr, NEMBD - 32, half);
  mac_group(acc, a0, b0);
  mac_group(acc, a1, b1);

  // C-layout: lane holds column n (+j*16), rows r + 8*half.
#pragma unroll
  for (int j = 0; j < 4; ++j) {
    int o_ = colGrp * 64 + j * 16 + n;
    int hh = o_ >> 6, dd = o_ & 63;
    float bj = bias[o_];
#pragma unroll
    for (int r = 0; r < 8; ++r) {
      int gr = rowTile * 16 + r + 8 * half;
      int bb = gr >> 11, t = gr & 2047;
      float y = acc[j][r] + bj;
      if (which == 0) {
        q[((size_t)(bb * NHEAD + hh) * TT + t) * HDIM + dd] =
            (__bf16)(y * 0.125f);                      // fold 1/sqrt(64)
      } else if (which == 1) {
        kmat[((size_t)(bb * NHEAD + hh) * TT + t) * HDIM + dd] = (__bf16)y;
      } else {
        vt[((size_t)(bb * NHEAD + hh) * HDIM + dd) * TT + t] = (__bf16)y;
      }
    }
  }
}

// ----------------------------------------------------- causal flash attention
// grid (32, 32), block 128 (4 waves, each an independent 16-row query tile).
// Tile-max online softmax (scores are ~N(0,1): tile max vs row max differ by a
// few units, far from expf underflow). Row sums come from an extra WMMA with an
// all-ones B, whose C-layout matches l_run[8] exactly.
__global__ void __launch_bounds__(128) attn_kernel(
    const __bf16* __restrict__ qb, const __bf16* __restrict__ kb,
    const __bf16* __restrict__ vtb, __bf16* __restrict__ aob) {
  __shared__ __bf16 PT[4][32 * 16];   // per-wave P^T tile: [key 0..31][row 0..15]

  const int lane = threadIdx.x & 31;
  const int wid  = threadIdx.x >> 5;
  const int half = lane >> 4;
  const int n    = lane & 15;
  const int qt   = blockIdx.x * 4 + wid;   // 0..127 query tile
  const int bh   = blockIdx.y;             // 0..31
  const int bi   = bh >> 4, h = bh & 15;

  const __bf16* qhead  = qb  + (size_t)bh * TT * HDIM;
  const __bf16* khead  = kb  + (size_t)bh * TT * HDIM;
  const __bf16* vthead = vtb + (size_t)bh * HDIM * TT;
  __bf16* lds = PT[wid];

  const int qrow = qt * 16 + n;
  v16bf aq0 = load_frag_row(qhead + (size_t)qrow * HDIM, 0, half);
  v16bf aq1 = load_frag_row(qhead + (size_t)qrow * HDIM, 32, half);

  FragH onesF;
#pragma unroll
  for (int e = 0; e < 16; ++e) onesF.h[e] = (__bf16)1.0f;
  const v16bf ones = onesF.v;

  v8f O[4] = {};
  float m_run = -__builtin_inff();
  float l_run[8];
#pragma unroll
  for (int r = 0; r < 8; ++r) l_run[r] = 0.f;

  const int nkb = (qt * 16 + 16 + 31) / 32;   // causal: skip fully-masked blocks
  for (int blk = 0; blk < nkb; ++blk) {
    const int kbase = blk * 32;

    // Issue all global fragment loads up front: 4 K-fragments for the score
    // tiles and 4 V-fragments for P@V (consumed after the softmax VALU work).
    const __bf16* kr0 = khead + (size_t)(kbase + n) * HDIM;
    const __bf16* kr1 = khead + (size_t)(kbase + 16 + n) * HDIM;
    v16bf bk00 = load_frag_row(kr0, 0, half);
    v16bf bk01 = load_frag_row(kr0, 32, half);
    v16bf bk10 = load_frag_row(kr1, 0, half);
    v16bf bk11 = load_frag_row(kr1, 32, half);
    v16bf bv[4];
#pragma unroll
    for (int j = 0; j < 4; ++j)
      bv[j] = load_frag_row(vthead + (size_t)(j * 16 + n) * TT, kbase, half);

    if (blk + 1 < nkb) {               // global_prefetch_b8 next key block
      __builtin_prefetch(kr0 + 32 * HDIM, 0, 0);
      __builtin_prefetch(kr1 + 32 * HDIM, 0, 0);
      __builtin_prefetch(vthead + (size_t)n * TT + kbase + 32, 0, 0);
    }

    // S = (q/sqrt(d)) @ K^T for two 16-key tiles, K-dim = 64 (2 chunks).
    v8f s0 = {}, s1 = {};
    s0 = wmma_bf16(aq0, bk00, s0);
    s0 = wmma_bf16(aq1, bk01, s0);
    s1 = wmma_bf16(aq0, bk10, s1);
    s1 = wmma_bf16(aq1, bk11, s1);

    // Causal mask + tile-wide running max.
    const int kc0 = kbase + n, kc1 = kbase + 16 + n;
    float v0[8], v1[8];
    float mx = -__builtin_inff();
#pragma unroll
    for (int r = 0; r < 8; ++r) {
      int qr = qt * 16 + r + 8 * half;
      v0[r] = (kc0 <= qr) ? s0[r] : -__builtin_inff();
      v1[r] = (kc1 <= qr) ? s1[r] : -__builtin_inff();
      mx = fmaxf(mx, fmaxf(v0[r], v1[r]));
    }
#pragma unroll
    for (int off = 1; off < 32; off <<= 1)
      mx = fmaxf(mx, __shfl_xor(mx, off, 32));
    const float mnew  = fmaxf(m_run, mx);
    const float alpha = __expf(m_run - mnew);
    m_run = mnew;

    // P = exp(S - m): lane holds a column -> store P^T column-contiguous
    // (one ds_store_b128 per 16-key tile).
    union { __bf16 h[8]; uint4 u; } w0, w1;
#pragma unroll
    for (int r = 0; r < 8; ++r) {
      w0.h[r] = (__bf16)__expf(v0[r] - mnew);
      w1.h[r] = (__bf16)__expf(v1[r] - mnew);
    }
    *reinterpret_cast<uint4*>(&lds[n * 16 + 8 * half])        = w0.u;
    *reinterpret_cast<uint4*>(&lds[(16 + n) * 16 + 8 * half]) = w1.u;
    asm volatile("s_wait_dscnt 0" ::: "memory");   // per-wave LDS visibility

    // Gather A-fragment of P (lane = row m = n, elements over key k).
    FragH fa;
#pragma unroll
    for (int e = 0; e < 16; ++e) {
      int vreg = e >> 1;
      int kk = ((vreg >> 2) << 4) + half * 8 + ((vreg & 3) << 1) + (e & 1);
      fa.h[e] = lds[kk * 16 + n];
    }
    const v16bf ap = fa.v;

    // Rescale running state; row sums via the matrix pipe: P @ ones.
#pragma unroll
    for (int j = 0; j < 4; ++j)
#pragma unroll
      for (int r = 0; r < 8; ++r) O[j][r] *= alpha;
    v8f lsum = {};
    lsum = wmma_bf16(ap, ones, lsum);
#pragma unroll
    for (int r = 0; r < 8; ++r) l_run[r] = l_run[r] * alpha + lsum[r];

    // O += P @ V  (B-fragment = rows of V^T, contiguous over t).
#pragma unroll
    for (int j = 0; j < 4; ++j) O[j] = wmma_bf16(ap, bv[j], O[j]);
  }

  // Normalize and scatter to [B, T, H*D] (bf16 for the output projection).
#pragma unroll
  for (int r = 0; r < 8; ++r) {
    int t = qt * 16 + r + 8 * half;
    float inv_l = 1.0f / l_run[r];
#pragma unroll
    for (int j = 0; j < 4; ++j) {
      aob[(size_t)(bi * TT + t) * NEMBD + h * HDIM + j * 16 + n] =
          (__bf16)(O[j][r] * inv_l);
    }
  }
}

// ------------------------------------------------------------ out projection
__global__ void __launch_bounds__(128) proj_gemm(
    const __bf16* __restrict__ A, const __bf16* __restrict__ W,
    const float* __restrict__ bias, float* __restrict__ out) {
  const int lane = threadIdx.x & 31;
  const int wid  = threadIdx.x >> 5;
  const int half = lane >> 4;
  const int n    = lane & 15;
  const int rowTile = blockIdx.y * 4 + wid;
  const int colGrp  = blockIdx.x;

  const __bf16* arow = A + (size_t)(rowTile * 16 + n) * NEMBD;
  const __bf16* wr[4];
#pragma unroll
  for (int j = 0; j < 4; ++j)
    wr[j] = W + (size_t)(colGrp * 64 + j * 16 + n) * NEMBD;

  v8f acc[4] = {};
  v16bf a0, a1, b0[4], b1[4];
  load_group(a0, b0, arow, wr, 0, half);
  for (int k0 = 32; k0 < NEMBD - 32; k0 += 64) {
    load_group(a1, b1, arow, wr, k0, half);
    mac_group(acc, a0, b0);
    load_group(a0, b0, arow, wr, k0 + 32, half);
    mac_group(acc, a1, b1);
  }
  load_group(a1, b1, arow, wr, NEMBD - 32, half);
  mac_group(acc, a0, b0);
  mac_group(acc, a1, b1);

#pragma unroll
  for (int j = 0; j < 4; ++j) {
    int o_ = colGrp * 64 + j * 16 + n;
    float bj = bias[o_];
#pragma unroll
    for (int r = 0; r < 8; ++r) {
      int gr = rowTile * 16 + r + 8 * half;
      out[(size_t)gr * NEMBD + o_] = acc[j][r] + bj;
    }
  }
}

// ---------------------------------------------------------------------------
extern "C" void kernel_launch(void* const* d_in, const int* in_sizes, int n_in,
                              void* d_out, int out_size, void* d_ws,
                              size_t ws_size, hipStream_t stream) {
  const float* x  = (const float*)d_in[0];
  const float* Wq = (const float*)d_in[1];
  const float* bq = (const float*)d_in[2];
  const float* Wk = (const float*)d_in[3];
  const float* bk = (const float*)d_in[4];
  const float* Wv = (const float*)d_in[5];
  const float* bv = (const float*)d_in[6];
  const float* Wp = (const float*)d_in[7];
  const float* bp = (const float*)d_in[8];

  __bf16* ws  = reinterpret_cast<__bf16*>(d_ws);
  __bf16* xb  = ws;                                  // [4096,1024] bf16
  __bf16* wqb = xb  + (size_t)BT * NEMBD;
  __bf16* wkb = wqb + (size_t)NEMBD * NEMBD;
  __bf16* wvb = wkb + (size_t)NEMBD * NEMBD;
  __bf16* wpb = wvb + (size_t)NEMBD * NEMBD;
  __bf16* qb  = wpb + (size_t)NEMBD * NEMBD;         // [B,H,T,D]
  __bf16* kb  = qb  + (size_t)BB * NHEAD * TT * HDIM;
  __bf16* vtb = kb  + (size_t)BB * NHEAD * TT * HDIM;// [B,H,D,T]
  __bf16* aob = xb;   // x is dead after qkv_gemm -> reuse for attention output

  const int xw4 = BT * NEMBD / 4;       // 1,048,576 float4s
  const int ww4 = NEMBD * NEMBD / 4;    //   262,144 float4s
  cvt_f32_bf16<<<(xw4 + 255) / 256, 256, 0, stream>>>(x,  xb,  xw4);
  cvt_f32_bf16<<<(ww4 + 255) / 256, 256, 0, stream>>>(Wq, wqb, ww4);
  cvt_f32_bf16<<<(ww4 + 255) / 256, 256, 0, stream>>>(Wk, wkb, ww4);
  cvt_f32_bf16<<<(ww4 + 255) / 256, 256, 0, stream>>>(Wv, wvb, ww4);
  cvt_f32_bf16<<<(ww4 + 255) / 256, 256, 0, stream>>>(Wp, wpb, ww4);

  qkv_gemm<<<dim3(16, 64, 3), 128, 0, stream>>>(xb, wqb, wkb, wvb,
                                                bq, bk, bv, qb, kb, vtb);
  attn_kernel<<<dim3(32, 32), 128, 0, stream>>>(qb, kb, vtb, aob);
  proj_gemm<<<dim3(16, 64), 128, 0, stream>>>(aob, wpb, bp, (float*)d_out);
}
